// RelativeMultiHeadSelfAttention_43310450212869
// MI455X (gfx1250) — matrix operand order repaired and checked by reading, not compiled, not run on hardware
//
#include <hip/hip_runtime.h>
#include <hip/hip_bf16.h>

// ---------------------------------------------------------------------------
// Relative-position MHSA for MI455X (gfx1250, wave32, WMMA bf16 path).
// All five einsums + 4 projections run on v_wmma_f32_16x16x32_bf16.
// Projection GEMMs use a workgroup-cooperative 3-deep async global->LDS
// pipeline (ASYNCcnt); attention stages K chunks with double-buffered async
// copies.
// ---------------------------------------------------------------------------

typedef __bf16 bf16_t;
typedef __bf16 v16bf __attribute__((ext_vector_type(16)));
typedef float  v8f   __attribute__((ext_vector_type(8)));
typedef unsigned int u32x4 __attribute__((ext_vector_type(4)));

#define DEV static __device__ __forceinline__

// ---------------- problem constants ----------------
constexpr int Bx = 4, Sx = 1024, Dx = 1024, Hx = 16, HDx = 64;
constexpr int ML = 1024;          // MAX_LEN
constexpr int Tx = 2 * ML + 1;    // 2049 relative-embedding rows
constexpr int TP = 2064;          // padded stride (multiple of 16)

constexpr size_t alup(size_t x) { return (x + 255) & ~size_t(255); }
// workspace layout (bytes)
constexpr size_t OFF_XB  = 0;                                         // x bf16 [4096,1024]
constexpr size_t OFF_WT  = alup(OFF_XB  + size_t(Bx * Sx) * Dx * 2);  // WqT,WkT,WvT,WoT bf16
constexpr size_t OFF_RKB = alup(OFF_WT  + size_t(4) * Dx * Dx * 2);   // rel_k bf16 [2049,64]
constexpr size_t OFF_RVT = alup(OFF_RKB + size_t(Tx) * HDx * 2);      // rel_vT bf16 [64,2064]
constexpr size_t OFF_QB  = alup(OFF_RVT + size_t(HDx) * TP * 2);      // Q bf16 [64,1024,64]
constexpr size_t OFF_KB  = alup(OFF_QB  + size_t(Bx * Hx) * Sx * HDx * 2);
constexpr size_t OFF_VT  = alup(OFF_KB  + size_t(Bx * Hx) * Sx * HDx * 2); // V^T [64,64,1024]
constexpr size_t OFF_AB  = alup(OFF_VT  + size_t(Bx * Hx) * HDx * Sx * 2); // attn bf16

// ---------------- WMMA helpers ----------------
DEV v8f wmma_bf16(v16bf a, v16bf b, v8f c) {
  return __builtin_amdgcn_wmma_f32_16x16x32_bf16(false, a, false, b, (short)0, c,
                                                 false, false);
}

// A-operand (16x32, M x K) from row-major [*, ld]: lane holds row (lane&15),
// K chunks {8*hi..8*hi+7} and {16+8*hi..16+8*hi+7}  (ISA 16-bit A 16x32 table).
DEV v16bf load_A(const bf16_t* p, int ld, int lane) {
  const int row  = lane & 15;
  const int koff = (lane >> 4) << 3;
  union { v16bf v; u32x4 u[2]; } r;
  r.u[0] = *(const u32x4*)(p + row * ld + koff);
  r.u[1] = *(const u32x4*)(p + row * ld + 16 + koff);
  return r.v;
}

// B-operand (32x16, K x N) supplied as B^T row-major [N, ld] (K contiguous):
// lane holds column (lane&15), K chunks {8*hi..8*hi+7} and {16+8*hi..16+8*hi+7}.
DEV v16bf load_B(const bf16_t* p, int ld, int lane) {
  const int n   = lane & 15;
  const int kb_ = (lane >> 4) << 3;
  const bf16_t* q = p + n * ld + kb_;
  union { v16bf v; u32x4 u[2]; } r;
  r.u[0] = *(const u32x4*)(q);
  r.u[1] = *(const u32x4*)(q + 16);
  return r.v;
}

DEV float rmax16(float v) {
  #pragma unroll
  for (int m = 1; m < 16; m <<= 1) v = fmaxf(v, __shfl_xor(v, m, 32));
  return v;
}
DEV float rsum16(float v) {
  #pragma unroll
  for (int m = 1; m < 16; m <<= 1) v += __shfl_xor(v, m, 32);
  return v;
}

// Wave-private LDS produce->consume fence (LDS ops are in-order per wave;
// only cross-lane visibility needs an explicit DScnt wait).
DEV void lds_wait() { asm volatile("s_wait_dscnt 0x0" ::: "memory"); }

// ---- CDNA5 async global->LDS copy (tracked by ASYNCcnt, in-order) ----
DEV void async_copy_b128(unsigned lds_off, const void* gptr) {
  asm volatile("global_load_async_to_lds_b128 %0, %1, off"
               :: "v"(lds_off), "v"(gptr) : "memory");
}
DEV void async_wait_le8() { asm volatile("s_wait_asynccnt 0x8" ::: "memory"); }
DEV void async_wait_le3() { asm volatile("s_wait_asynccnt 0x3" ::: "memory"); }
DEV void async_wait_0()   { asm volatile("s_wait_asynccnt 0x0" ::: "memory"); }

// Stage this wave's private A tile (16 rows x 32 k = 16 x 64B): 2 async ops.
DEV void stage_a16x32(const bf16_t* Ap /*at (m0,k0), ld=Dx*/, unsigned lds_off,
                      int lane) {
  const int seg = lane & 3, r0 = lane >> 2;  // 4 x 16B segments per row
  #pragma unroll
  for (int it = 0; it < 2; ++it) {
    const int r = r0 + it * 8;
    async_copy_b128(lds_off + r * 64 + seg * 16,
                    (const char*)(Ap + size_t(r) * Dx) + seg * 16);
  }
}
// Stage this wave's 8-row share of the WG-shared B slab (8 x 64B): 1 async op.
DEV void stage_b8x32(const bf16_t* Wp /*at (n0+wid*8,k0), ld=Dx*/,
                     unsigned lds_off /*already + wid*8*64*/, int lane) {
  const int seg = lane & 3, r = lane >> 2;
  async_copy_b128(lds_off + r * 64 + seg * 16,
                  (const char*)(Wp + size_t(r) * Dx) + seg * 16);
}
// Contiguous 4KB K chunk (32 rows x 64 bf16): 8 async ops.
DEV void stage_k_chunk(const bf16_t* gsrc, unsigned lds_off, int lane) {
  const char* gp = (const char*)gsrc + lane * 16;
  const unsigned lp = lds_off + lane * 16;
  #pragma unroll
  for (int it = 0; it < 8; ++it)
    async_copy_b128(lp + it * 512, gp + it * 512);
}

// ---------------------------------------------------------------------------
// Kernel 1: fp32 -> bf16 conversions / transposes
// ---------------------------------------------------------------------------
__global__ void prep_kernel(const float* __restrict__ x,
                            const float* __restrict__ Wq, const float* __restrict__ Wk,
                            const float* __restrict__ Wv, const float* __restrict__ Wo,
                            const float* __restrict__ rk, const float* __restrict__ rv,
                            bf16_t* __restrict__ xb, bf16_t* __restrict__ wT,
                            bf16_t* __restrict__ rkb, bf16_t* __restrict__ rvT) {
  const int NX = Bx * Sx * Dx;  // 4M, largest region
  for (int i = blockIdx.x * blockDim.x + threadIdx.x; i < NX;
       i += gridDim.x * blockDim.x) {
    xb[i] = (bf16_t)x[i];
    if (i < Dx * Dx) {  // transpose weights: wT[n*Dx+k] = W[k*Dx+n]
      int k = i >> 10, n = i & (Dx - 1);
      size_t o = size_t(n) * Dx + k;
      wT[o]                        = (bf16_t)Wq[i];
      wT[size_t(1) * Dx * Dx + o]  = (bf16_t)Wk[i];
      wT[size_t(2) * Dx * Dx + o]  = (bf16_t)Wv[i];
      wT[size_t(3) * Dx * Dx + o]  = (bf16_t)Wo[i];
    }
    if (i < Tx * HDx) rkb[i] = (bf16_t)rk[i];
    if (i < HDx * TP) {  // rel_v transposed + zero padded
      int d = i / TP, t = i - d * TP;
      rvT[i] = (t < Tx) ? (bf16_t)rv[t * HDx + d] : (bf16_t)0.f;
    }
  }
}

// ---------------------------------------------------------------------------
// Cooperative pipelined GEMM body: one wave computes a 16x64 fp32 tile.
// 3 LDS buffers, prefetch distance 2, exactly 3 async ops per wave per stage
// (2 A + 1 B-share) => s_wait_asynccnt 0x3 makes stage k ready while stage
// k+1 is still in flight.  B slab is shared by all 8 waves (8x less traffic).
// ---------------------------------------------------------------------------
#define GEMM_PIPE_BODY(Ap, Wp, ASL, BSL, acc)                                   \
  {                                                                             \
    unsigned aoff[3], boff[3];                                                  \
    _Pragma("unroll")                                                           \
    for (int b = 0; b < 3; ++b) {                                               \
      aoff[b] = (unsigned)(uintptr_t)(const void*)&ASL[b][wid][0];              \
      boff[b] = (unsigned)(uintptr_t)(const void*)&BSL[b][0] + wid * 8 * 64;    \
    }                                                                           \
    stage_a16x32(Ap, aoff[0], lane);                                            \
    stage_b8x32(Wp, boff[0], lane);                                             \
    stage_a16x32(Ap + 32, aoff[1], lane);                                       \
    stage_b8x32(Wp + 32, boff[1], lane);                                        \
    _Pragma("unroll 1")                                                         \
    for (int k0 = 0, st = 0; k0 < Dx; k0 += 32, ++st) {                         \
      const int buf = st - (st / 3) * 3;                                        \
      async_wait_le3();   /* own stage-st ops landed (in-order) */              \
      __syncthreads();    /* everyone's B shares landed; prior reads done */    \
      v16bf a = load_A(&ASL[buf][wid][0], 32, lane);                            \
      _Pragma("unroll")                                                         \
      for (int nt = 0; nt < 4; ++nt) {                                          \
        v16bf b = load_B(&BSL[buf][0] + nt * 16 * 32, 32, lane);                \
        acc[nt] = wmma_bf16(a, b, acc[nt]);                                     \
      }                                                                         \
      if (k0 + 64 < Dx) {                                                       \
        const int nb = (st + 2) - ((st + 2) / 3) * 3;                           \
        stage_a16x32(Ap + k0 + 64, aoff[nb], lane);                             \
        stage_b8x32(Wp + k0 + 64, boff[nb], lane);                              \
      }                                                                         \
    }                                                                           \
  }

// ---------------------------------------------------------------------------
// Kernel 2: QKV projections  Y = x @ W + b
// grid (16, 32, 3): 64 cols x 128 rows per WG, z selects q/k/v.
// Q,K stored [bh, s, d]; V stored transposed [bh, d, s].
// ---------------------------------------------------------------------------
__global__ void __launch_bounds__(256)
qkv_kernel(const bf16_t* __restrict__ xb, const bf16_t* __restrict__ wT,
           const float* __restrict__ bq, const float* __restrict__ bk,
           const float* __restrict__ bv, bf16_t* __restrict__ qb,
           bf16_t* __restrict__ kb, bf16_t* __restrict__ vt) {
  __shared__ __align__(16) bf16_t asl[3][8][16 * 32];  // 24 KB
  __shared__ __align__(16) bf16_t bsl[3][64 * 32];     // 12 KB

  const int lane = threadIdx.x & 31, wid = threadIdx.x >> 5;
  const int z = blockIdx.z;
  const bf16_t* W   = wT + size_t(z) * Dx * Dx;
  const float* bias = (z == 0) ? bq : (z == 1) ? bk : bv;
  const int m0 = blockIdx.y * 128 + wid * 16;
  const int n0 = blockIdx.x * 64;
  const bf16_t* Ap = xb + size_t(m0) * Dx;
  const bf16_t* Wp = W + size_t(n0 + wid * 8) * Dx;  // this wave's B share

  v8f acc[4] = {};
  GEMM_PIPE_BODY(Ap, Wp, asl, bsl, acc)

  const int hi = lane >> 4, nl = lane & 15;
  #pragma unroll
  for (int nt = 0; nt < 4; ++nt) {
    const int col = n0 + nt * 16 + nl;
    const float bz = bias[col];
    const int h = col >> 6, d = col & 63;
    #pragma unroll
    for (int r = 0; r < 8; ++r) {
      const int row = m0 + r + 8 * hi;
      const int b_ = row >> 10, s = row & (Sx - 1);
      bf16_t val = (bf16_t)(acc[nt][r] + bz);
      if (z == 2)      vt[(size_t((b_ * Hx + h) * HDx + d)) * Sx + s] = val;
      else if (z == 0) qb[(size_t((b_ * Hx + h) * Sx + s)) * HDx + d] = val;
      else             kb[(size_t((b_ * Hx + h) * Sx + s)) * HDx + d] = val;
    }
  }
}

// ---------------------------------------------------------------------------
// Kernel 3: flash attention with relative positions.
// grid (8, 64): blockIdx.y = (b*16+h), blockIdx.x = 128-query super-tile.
// Each of the 8 waves owns 16 queries (Q resident in regs), streams 32 keys
// per iteration. K chunks are staged through double-buffered LDS via async
// global->LDS copies; next chunk is fetched while current chunk's 22 WMMAs
// (content + rel-k skew + P@V + P2@rel_v) execute.
// ---------------------------------------------------------------------------
__global__ void __launch_bounds__(256)
attn_kernel(const bf16_t* __restrict__ qb, const bf16_t* __restrict__ kb,
            const bf16_t* __restrict__ vt, const bf16_t* __restrict__ rkb,
            const bf16_t* __restrict__ rvT, bf16_t* __restrict__ ab) {
  __shared__ __align__(16) float  rqs[8][16 * 48];       // Rq diag window, fp32
  __shared__ __align__(16) bf16_t pds[8][16 * 32];       // dense probs (PV A)
  __shared__ __align__(16) bf16_t p2s[8][2][16 * 32];    // skewed probs (relV A)
  __shared__ __align__(16) bf16_t kst[8][2][32 * HDx];   // K chunk double buffer

  const int lane = threadIdx.x & 31, w = threadIdx.x >> 5;
  const int hi = lane >> 4, nl = lane & 15;
  const int bh = blockIdx.y;
  const int i0 = blockIdx.x * 128 + w * 16;

  const bf16_t* Q = qb + (size_t(bh) * Sx + i0) * HDx;
  const bf16_t* K = kb + size_t(bh) * Sx * HDx;
  const bf16_t* V = vt + size_t(bh) * HDx * Sx;

  float*  rq     = rqs[w];
  bf16_t* pde    = pds[w];
  bf16_t* p2base = &p2s[w][0][0];
  const unsigned kst_off[2] = {
      (unsigned)(uintptr_t)(const void*)&kst[w][0][0],
      (unsigned)(uintptr_t)(const void*)&kst[w][1][0]};

  // Q tile (16x64) resident as two A-operands
  v16bf qa0 = load_A(Q, HDx, lane);
  v16bf qa1 = load_A(Q + 32, HDx, lane);

  v8f o[4] = {};
  float Mrow[8], Lrow[8];
  #pragma unroll
  for (int r = 0; r < 8; ++r) { Mrow[r] = -1e30f; Lrow[r] = 0.f; }

  // prologue: stage chunk 0
  stage_k_chunk(K, kst_off[0], lane);

  #pragma unroll 1
  for (int j0 = 0; j0 < Sx; j0 += 32) {
    const int buf = (j0 >> 5) & 1;
    const bf16_t* kbuf = &kst[w][buf][0];
    if (j0 + 32 < Sx) {  // fetch next chunk while we compute on this one
      stage_k_chunk(K + size_t(j0 + 32) * HDx, kst_off[buf ^ 1], lane);
      __builtin_prefetch(V + (j0 + 32), 0, 1);
      async_wait_le8();  // in-order: <=8 outstanding => current chunk landed
    } else {
      async_wait_0();
    }
    const int c0 = i0 - j0 + ML;  // diag center, multiple of 16, in [32,2032]

    // ---- content scores: Sc = Q @ K^T  (B-operands from LDS-staged K) ----
    v8f sc[2] = {};
    {
      v16bf kb00 = load_B(kbuf + 0 * HDx,       HDx, lane);
      v16bf kb01 = load_B(kbuf + 0 * HDx + 32,  HDx, lane);
      v16bf kb10 = load_B(kbuf + 16 * HDx,      HDx, lane);
      v16bf kb11 = load_B(kbuf + 16 * HDx + 32, HDx, lane);
      sc[0] = wmma_bf16(qa0, kb00, sc[0]);
      sc[0] = wmma_bf16(qa1, kb01, sc[0]);
      sc[1] = wmma_bf16(qa0, kb10, sc[1]);
      sc[1] = wmma_bf16(qa1, kb11, sc[1]);
    }

    // ---- relative-key: Rq[m,t] = q[m,:]·rel_k_emb[t,:], t in [c0-32, c0+15]
    v16bf rb[6];
    #pragma unroll
    for (int tt = 0; tt < 3; ++tt) {
      const int t0 = c0 - 32 + tt * 16;  // in [0, 2032]
      rb[2 * tt]     = load_B(rkb + size_t(t0) * HDx,      HDx, lane);
      rb[2 * tt + 1] = load_B(rkb + size_t(t0) * HDx + 32, HDx, lane);
    }
    #pragma unroll
    for (int tt = 0; tt < 3; ++tt) {
      v8f rt = {};
      rt = wmma_bf16(qa0, rb[2 * tt], rt);
      rt = wmma_bf16(qa1, rb[2 * tt + 1], rt);
      #pragma unroll
      for (int r = 0; r < 8; ++r)
        rq[(r + 8 * hi) * 48 + tt * 16 + nl] = rt[r];
    }
    lds_wait();
    // diagonal gather: score(m, j0+16s+n) += Rq[m, c0-16s+m-n]
    #pragma unroll
    for (int s = 0; s < 2; ++s)
      #pragma unroll
      for (int r = 0; r < 8; ++r) {
        const int m = r + 8 * hi;
        sc[s][r] += rq[m * 48 + (32 - 16 * s + m - nl)];
      }

    // ---- online softmax (unscaled logits, per reference) ----
    float scale[8];
    #pragma unroll
    for (int r = 0; r < 8; ++r) {
      float mx = rmax16(fmaxf(sc[0][r], sc[1][r]));
      float nm = fmaxf(Mrow[r], mx);
      scale[r] = __expf(Mrow[r] - nm);
      Mrow[r] = nm;
      float p0 = __expf(sc[0][r] - nm);
      float p1 = __expf(sc[1][r] - nm);
      sc[0][r] = p0; sc[1][r] = p1;
      Lrow[r] = Lrow[r] * scale[r] + rsum16(p0 + p1);
    }
    #pragma unroll
    for (int dt = 0; dt < 4; ++dt)
      #pragma unroll
      for (int r = 0; r < 8; ++r) o[dt][r] *= scale[r];

    // ---- zero skew buffers, then scatter probs (dense + skewed) ----
    {
      u32x4 z4 = {0u, 0u, 0u, 0u};
      u32x4* zz = (u32x4*)p2base;
      #pragma unroll
      for (int t = 0; t < 4; ++t) zz[lane + 32 * t] = z4;
    }
    #pragma unroll
    for (int s = 0; s < 2; ++s)
      #pragma unroll
      for (int r = 0; r < 8; ++r) {
        const int m = r + 8 * hi;
        bf16_t pv = (bf16_t)sc[s][r];
        pde[m * 32 + s * 16 + nl] = pv;                      // dense 16x32
        p2base[s * (16 * 32) + m * 32 + (m - nl + 16)] = pv; // skewed window
      }

    // issue V B-operand loads BEFORE the LDS fence so they fly during it
    v16bf vb[4];
    #pragma unroll
    for (int dt = 0; dt < 4; ++dt)
      vb[dt] = load_B(V + size_t(dt * 16) * Sx + j0, Sx, lane);
    lds_wait();

    // ---- O += P @ V ----
    v16bf pa = load_A(pde, 32, lane);
    #pragma unroll
    for (int dt = 0; dt < 4; ++dt) o[dt] = wmma_bf16(pa, vb[dt], o[dt]);

    // ---- O += P2 @ rel_v_emb (t-window base c0-16s-16, 16-aligned) ----
    #pragma unroll
    for (int s = 0; s < 2; ++s) {
      const int tb = c0 - 16 * s - 16;  // in [0, 2016]
      v16bf pa2 = load_A(p2base + s * (16 * 32), 32, lane);
      #pragma unroll
      for (int dt = 0; dt < 4; ++dt) {
        v16bf bv_ = load_B(rvT + size_t(dt * 16) * TP + tb, TP, lane);
        o[dt] = wmma_bf16(pa2, bv_, o[dt]);
      }
    }
  }

  // ---- normalize and store attn as bf16 [b, s, h*64+d] for out-projection ----
  const int b_ = bh >> 4, h = bh & (Hx - 1);
  #pragma unroll
  for (int r = 0; r < 8; ++r) Lrow[r] = 1.f / Lrow[r];
  #pragma unroll
  for (int dt = 0; dt < 4; ++dt)
    #pragma unroll
    for (int r = 0; r < 8; ++r) {
      const int i = i0 + r + 8 * hi;
      const int d = dt * 16 + nl;
      ab[(size_t(b_) * Sx + i) * Dx + h * HDx + d] = (bf16_t)(o[dt][r] * Lrow[r]);
    }
}

// ---------------------------------------------------------------------------
// Kernel 4: output projection  out = attn @ Wo + bo  (same pipelined GEMM)
// ---------------------------------------------------------------------------
__global__ void __launch_bounds__(256)
proj_kernel(const bf16_t* __restrict__ ab, const bf16_t* __restrict__ woT,
            const float* __restrict__ bo, float* __restrict__ out) {
  __shared__ __align__(16) bf16_t asl[3][8][16 * 32];
  __shared__ __align__(16) bf16_t bsl[3][64 * 32];

  const int lane = threadIdx.x & 31, wid = threadIdx.x >> 5;
  const int m0 = blockIdx.y * 128 + wid * 16;
  const int n0 = blockIdx.x * 64;
  const bf16_t* Ap = ab + size_t(m0) * Dx;
  const bf16_t* Wp = woT + size_t(n0 + wid * 8) * Dx;

  v8f acc[4] = {};
  GEMM_PIPE_BODY(Ap, Wp, asl, bsl, acc)

  const int hi = lane >> 4, nl = lane & 15;
  #pragma unroll
  for (int nt = 0; nt < 4; ++nt) {
    const int col = n0 + nt * 16 + nl;
    const float bz = bo[col];
    #pragma unroll
    for (int r = 0; r < 8; ++r) {
      const int row = m0 + r + 8 * hi;
      out[size_t(row) * Dx + col] = acc[nt][r] + bz;
    }
  }
}

// ---------------------------------------------------------------------------
extern "C" void kernel_launch(void* const* d_in, const int* in_sizes, int n_in,
                              void* d_out, int out_size, void* d_ws, size_t ws_size,
                              hipStream_t stream) {
  const float* x  = (const float*)d_in[0];
  const float* Wq = (const float*)d_in[1];
  const float* bq = (const float*)d_in[2];
  const float* Wk = (const float*)d_in[3];
  const float* bk = (const float*)d_in[4];
  const float* Wv = (const float*)d_in[5];
  const float* bv = (const float*)d_in[6];
  const float* Wo = (const float*)d_in[7];
  const float* bo = (const float*)d_in[8];
  const float* rk = (const float*)d_in[9];
  const float* rv = (const float*)d_in[10];
  float* out = (float*)d_out;

  char* ws = (char*)d_ws;
  bf16_t* xb  = (bf16_t*)(ws + OFF_XB);
  bf16_t* wT  = (bf16_t*)(ws + OFF_WT);
  bf16_t* rkb = (bf16_t*)(ws + OFF_RKB);
  bf16_t* rvT = (bf16_t*)(ws + OFF_RVT);
  bf16_t* qb  = (bf16_t*)(ws + OFF_QB);
  bf16_t* kb  = (bf16_t*)(ws + OFF_KB);
  bf16_t* vt  = (bf16_t*)(ws + OFF_VT);
  bf16_t* abf = (bf16_t*)(ws + OFF_AB);

  prep_kernel<<<dim3(1024), 256, 0, stream>>>(x, Wq, Wk, Wv, Wo, rk, rv,
                                              xb, wT, rkb, rvT);
  qkv_kernel<<<dim3(16, 32, 3), 256, 0, stream>>>(xb, wT, bq, bk, bv, qb, kb, vt);
  attn_kernel<<<dim3(8, 64), 256, 0, stream>>>(qb, kb, vt, rkb, rvT, abf);
  proj_kernel<<<dim3(16, 32), 256, 0, stream>>>(abf, wT + size_t(3) * Dx * Dx, bo, out);

  (void)in_sizes; (void)n_in; (void)out_size; (void)ws_size;
}
